// SPSAttention_62517543961223
// MI455X (gfx1250) — matrix operand order repaired and checked by reading, not compiled, not run on hardware
//
#include <hip/hip_runtime.h>
#include <hip/hip_bf16.h>

typedef __bf16 bf16_t;
typedef __attribute__((ext_vector_type(16))) __bf16 v16bf;
typedef __attribute__((ext_vector_type(8)))  __bf16 v8bf;
typedef __attribute__((ext_vector_type(8)))  float  v8f;

// Global-address-space views so fragment loads lower to global_load_b128
// (LOADcnt only) rather than flat_load_b128 (LOADcnt + DScnt coupling).
typedef __attribute__((address_space(1))) const __bf16 gbf;
typedef __attribute__((address_space(1))) const v8bf  gv8bf;

#define BB 4
#define SS 2048
#define DD 1024
#define ROWS (BB * SS)      // 8192
#define QKV_LD (3 * DD)     // 3072
#define ATTN_SCALE 0.125f   // HEAD_DIM^-0.5 = 64^-0.5

// ---------------------------------------------------------------------------
// CDNA5 wave32 WMMA 16-bit operand layouts (ISA 7.12.2):
// A (16x32 MxK): lane l -> row m=l&15; half kh=l>>4 holds K = kh*8+[0..7]
//                and 16+kh*8+[0..7]  (two b128 chunks at +0 and +32 bytes)
// B (32x16 KxN) from row-major [N,K] storage: lane l -> col n=l&15;
//                half kh holds K = kh*8+[0..7] and 16+kh*8+[0..7] (b128 at +0 and +32 bytes)
// ---------------------------------------------------------------------------
static __device__ inline v16bf frag_a(gbf* p) {
  v8bf lo = *(gv8bf*)(p);
  v8bf hi = *(gv8bf*)(p + 16);
  return __builtin_shufflevector(lo, hi, 0,1,2,3,4,5,6,7,8,9,10,11,12,13,14,15);
}
static __device__ inline v16bf frag_b(gbf* p) {
  v8bf lo = *(gv8bf*)(p);
  v8bf hi = *(gv8bf*)(p + 16);
  return __builtin_shufflevector(lo, hi, 0,1,2,3,4,5,6,7,8,9,10,11,12,13,14,15);
}

// One wave computes a 64x64 f32 tile: C[m0.., n0..] = A[M,K] @ B[N,K]^T.
// Per-lane base addresses hoisted (in AS1); inner loop = b128 loads + 16 WMMA.
static __device__ inline void wmma_tile_64x64(const bf16_t* __restrict__ A, int lda,
                                              const bf16_t* __restrict__ Bt, int ldb,
                                              int m0, int n0, int K,
                                              v8f (&acc)[4][4]) {
  const int lane = threadIdx.x & 31;
  const int rl = lane & 15;
  const int kh = lane >> 4;
  gbf* ap[4];
  gbf* bp[4];
#pragma unroll
  for (int i = 0; i < 4; ++i)
    ap[i] = (gbf*)(A + (size_t)(m0 + 16 * i + rl) * lda + (kh << 3));
#pragma unroll
  for (int j = 0; j < 4; ++j)
    bp[j] = (gbf*)(Bt + (size_t)(n0 + 16 * j + rl) * ldb + (kh << 3));

  for (int k0 = 0; k0 < K; k0 += 32) {
    v16bf a[4], b[4];
#pragma unroll
    for (int i = 0; i < 4; ++i) { a[i] = frag_a(ap[i]); ap[i] += 32; }
#pragma unroll
    for (int j = 0; j < 4; ++j) { b[j] = frag_b(bp[j]); bp[j] += 32; }
#pragma unroll
    for (int i = 0; i < 4; ++i)
#pragma unroll
      for (int j = 0; j < 4; ++j)
        acc[i][j] = __builtin_amdgcn_wmma_f32_16x16x32_bf16(
            false, a[i], false, b[j], (short)0, acc[i][j], false, false);
  }
}

// C/D layout (ISA 7.12.2): lane l, vgpr r -> m = r + 8*(l>>4), n = l&15.
#define TILE_COORDS()                         \
  const int wave = threadIdx.x >> 5;          \
  const int lane = threadIdx.x & 31;          \
  const int half = lane >> 4;                 \
  const int nl   = lane & 15;                 \
  const int m0 = blockIdx.x * 256 + (wave & 3) * 64; \
  const int n0 = blockIdx.y * 128 + (wave >> 2) * 64;

// ---------------------------------------------------------------------------
// Elementwise convert f32 -> bf16
// ---------------------------------------------------------------------------
__global__ void __launch_bounds__(256)
k_f32_to_bf16(const float* __restrict__ in, bf16_t* __restrict__ out, int n) {
  int i = blockIdx.x * blockDim.x + threadIdx.x;
  if (i < n) out[i] = (bf16_t)in[i];
}

// in: [K,N] f32 row-major -> out: [N,K] bf16 row-major
__global__ void __launch_bounds__(256)
k_transpose_bf16(const float* __restrict__ in, bf16_t* __restrict__ out, int K, int N) {
  size_t idx = (size_t)blockIdx.x * blockDim.x + threadIdx.x;
  if (idx >= (size_t)K * N) return;
  int k = (int)(idx % K);
  int n = (int)(idx / K);
  out[idx] = (bf16_t)in[(size_t)k * N + n];
}

// ---------------------------------------------------------------------------
// qkv = x @ W_qkv + b_qkv   (M=8192, N=3072, K=1024), output bf16.
// Also side-stores V transposed per batch: Vt[b][d][t].
// ---------------------------------------------------------------------------
__global__ void __launch_bounds__(256, 1)
k_gemm_qkv(const bf16_t* __restrict__ X, const bf16_t* __restrict__ Wt,
           const float* __restrict__ bias,
           bf16_t* __restrict__ qkv, bf16_t* __restrict__ Vt) {
  TILE_COORDS();
  v8f acc[4][4] = {};
  wmma_tile_64x64(X, DD, Wt, DD, m0, n0, DD, acc);
#pragma unroll
  for (int i = 0; i < 4; ++i)
#pragma unroll
    for (int j = 0; j < 4; ++j) {
      const int n = n0 + j * 16 + nl;
      const float bv = bias[n];
#pragma unroll
      for (int r = 0; r < 8; ++r) {
        const int m = m0 + i * 16 + r + 8 * half;
        const float v = acc[i][j][r] + bv;
        qkv[(size_t)m * QKV_LD + n] = (bf16_t)v;
        if (n >= 2 * DD) {  // v-part: also store transposed [b][d][t]
          const int b = m >> 11, t = m & (SS - 1), d = n - 2 * DD;
          Vt[(((size_t)b * DD) + d) * SS + t] = (bf16_t)v;
        }
      }
    }
}

// ---------------------------------------------------------------------------
// scores[b] = sps_pro(scale * q_b @ k_b^T)   (per batch M=N=2048, K=1024), f32 out
// ---------------------------------------------------------------------------
__global__ void __launch_bounds__(256, 1)
k_scores(const bf16_t* __restrict__ qkv, float* __restrict__ S) {
  TILE_COORDS();
  const int b = blockIdx.z;
  const bf16_t* q = qkv + (size_t)b * SS * QKV_LD;          // q part
  const bf16_t* kmat = q + DD;                              // k part (same rows)
  v8f acc[4][4] = {};
  // B operand is k^T: B[d,t] = k[t,d] -> contiguous rows of k
  wmma_tile_64x64(q, QKV_LD, kmat, QKV_LD, m0, n0, DD, acc);
  float* Sb = S + (size_t)b * SS * SS;
#pragma unroll
  for (int i = 0; i < 4; ++i)
#pragma unroll
    for (int j = 0; j < 4; ++j)
#pragma unroll
      for (int r = 0; r < 8; ++r) {
        const int m = m0 + i * 16 + r + 8 * half;
        const int n = n0 + j * 16 + nl;
        float s = acc[i][j][r] * ATTN_SCALE;
        const float xn = 1.0f / (1.0f + __expf(-s));       // sigmoid
        const float mult = (xn > 0.95f) ? 2.0f
                         : (xn > 0.85f) ? 1.5f
                         : (xn < 0.10f) ? 0.1f
                         : (xn < 0.25f) ? 0.3f : 1.0f;
        Sb[(size_t)m * SS + n] = s * mult;
      }
}

// ---------------------------------------------------------------------------
// In-place row softmax over 2048 f32; rewrites the row's first 4KB as bf16 P.
// One 256-thread block per row; all reads complete before barriers, writes after.
// ---------------------------------------------------------------------------
__global__ void __launch_bounds__(256)
k_softmax(float* __restrict__ S) {
  float* srow = S + (size_t)blockIdx.x * SS;
  const int t = threadIdx.x;
  float v[8];
#pragma unroll
  for (int i = 0; i < 8; ++i) v[i] = srow[t + 256 * i];

  float mx = v[0];
#pragma unroll
  for (int i = 1; i < 8; ++i) mx = fmaxf(mx, v[i]);
#pragma unroll
  for (int off = 16; off > 0; off >>= 1) mx = fmaxf(mx, __shfl_xor(mx, off, 32));
  __shared__ float redm[8];
  if ((t & 31) == 0) redm[t >> 5] = mx;
  __syncthreads();
  float bm = redm[0];
#pragma unroll
  for (int w = 1; w < 8; ++w) bm = fmaxf(bm, redm[w]);

  float e[8], sum = 0.0f;
#pragma unroll
  for (int i = 0; i < 8; ++i) { e[i] = __expf(v[i] - bm); sum += e[i]; }
#pragma unroll
  for (int off = 16; off > 0; off >>= 1) sum += __shfl_xor(sum, off, 32);
  __shared__ float reds[8];
  if ((t & 31) == 0) reds[t >> 5] = sum;
  __syncthreads();
  float bs = 0.0f;
#pragma unroll
  for (int w = 0; w < 8; ++w) bs += reds[w];
  const float inv = 1.0f / bs;

  bf16_t* prow = (bf16_t*)srow;  // in-place: bf16 row occupies first half of f32 row
#pragma unroll
  for (int i = 0; i < 8; ++i) prow[t + 256 * i] = (bf16_t)(e[i] * inv);
}

// ---------------------------------------------------------------------------
// O[b] = P_b @ v_b   (per batch M=2048, N=1024, K=2048), bf16 out.
// P rows live at stride 4096 bf16 (first half of each 8KB f32 row).
// ---------------------------------------------------------------------------
__global__ void __launch_bounds__(256, 1)
k_gemm_pv(const float* __restrict__ Sbuf, const bf16_t* __restrict__ Vt,
          bf16_t* __restrict__ O) {
  TILE_COORDS();
  const int b = blockIdx.z;
  const bf16_t* P = (const bf16_t*)(Sbuf + (size_t)b * SS * SS);  // ld = 2*SS bf16
  const bf16_t* V = Vt + (size_t)b * DD * SS;                     // [d][t], ld = SS
  v8f acc[4][4] = {};
  wmma_tile_64x64(P, 2 * SS, V, SS, m0, n0, SS, acc);
#pragma unroll
  for (int i = 0; i < 4; ++i)
#pragma unroll
    for (int j = 0; j < 4; ++j)
#pragma unroll
      for (int r = 0; r < 8; ++r) {
        const int m = m0 + i * 16 + r + 8 * half;
        const int n = n0 + j * 16 + nl;
        O[((size_t)b * SS + m) * DD + n] = (bf16_t)acc[i][j][r];
      }
}

// ---------------------------------------------------------------------------
// out = O @ W_proj + b_proj   (M=8192, N=1024, K=1024), f32 out.
// ---------------------------------------------------------------------------
__global__ void __launch_bounds__(256, 1)
k_gemm_proj(const bf16_t* __restrict__ O, const bf16_t* __restrict__ Wt,
            const float* __restrict__ bias, float* __restrict__ out) {
  TILE_COORDS();
  v8f acc[4][4] = {};
  wmma_tile_64x64(O, DD, Wt, DD, m0, n0, DD, acc);
#pragma unroll
  for (int i = 0; i < 4; ++i)
#pragma unroll
    for (int j = 0; j < 4; ++j) {
      const int n = n0 + j * 16 + nl;
      const float bv = bias[n];
#pragma unroll
      for (int r = 0; r < 8; ++r) {
        const int m = m0 + i * 16 + r + 8 * half;
        out[(size_t)m * DD + n] = acc[i][j][r] + bv;
      }
    }
}

extern "C" void kernel_launch(void* const* d_in, const int* in_sizes, int n_in,
                              void* d_out, int out_size, void* d_ws, size_t ws_size,
                              hipStream_t stream) {
  const float* x      = (const float*)d_in[0];
  const float* W_qkv  = (const float*)d_in[1];
  const float* b_qkv  = (const float*)d_in[2];
  const float* W_proj = (const float*)d_in[3];
  const float* b_proj = (const float*)d_in[4];
  float* out = (float*)d_out;

  char* ws = (char*)d_ws;
  size_t off = 0;
  auto alloc = [&](size_t bytes) -> void* {
    void* p = ws + off;
    off += (bytes + 255) & ~(size_t)255;
    return p;
  };
  bf16_t* x_bf   = (bf16_t*)alloc((size_t)ROWS * DD * 2);        // 16 MiB
  bf16_t* WtQKV  = (bf16_t*)alloc((size_t)QKV_LD * DD * 2);      //  6 MiB
  bf16_t* WtProj = (bf16_t*)alloc((size_t)DD * DD * 2);          //  2 MiB
  bf16_t* qkv    = (bf16_t*)alloc((size_t)ROWS * QKV_LD * 2);    // 48 MiB
  bf16_t* Vt     = (bf16_t*)alloc((size_t)BB * DD * SS * 2);     // 16 MiB
  float*  Sc     = (float*) alloc((size_t)BB * SS * SS * 4);     // 64 MiB
  bf16_t* O_bf   = x_bf;  // x_bf is dead after k_gemm_qkv; reuse for attention output

  const int NX = ROWS * DD;  // 8388608
  k_f32_to_bf16 <<<(NX + 255) / 256, 256, 0, stream>>>(x, x_bf, NX);
  k_transpose_bf16<<<(QKV_LD * DD + 255) / 256, 256, 0, stream>>>(W_qkv, WtQKV, DD, QKV_LD);
  k_transpose_bf16<<<(DD * DD + 255) / 256, 256, 0, stream>>>(W_proj, WtProj, DD, DD);

  k_gemm_qkv <<<dim3(ROWS / 256, QKV_LD / 128), 256, 0, stream>>>(x_bf, WtQKV, b_qkv, qkv, Vt);
  k_scores   <<<dim3(SS / 256, SS / 128, BB), 256, 0, stream>>>(qkv, Sc);
  k_softmax  <<<ROWS, 256, 0, stream>>>(Sc);
  k_gemm_pv  <<<dim3(SS / 256, DD / 128, BB), 256, 0, stream>>>(Sc, Vt, O_bf);
  k_gemm_proj<<<dim3(ROWS / 256, DD / 128), 256, 0, stream>>>(O_bf, WtProj, b_proj, out);
}
